// ConvQRNN_27238682591833
// MI455X (gfx1250) — matrix operand order repaired and checked by reading, not compiled, not run on hardware
//
#include <hip/hip_runtime.h>

// ---------------------------------------------------------------------------
// Fused ConvQRNN for MI455X (gfx1250, wave32, WMMA)
//   X:(4,3,32,64,64) f32 -> conv3d(2x3x3, 256 out) -> peephole LSTM scan
//   out:(4,64,32,64,64) f32
// Block = (batch, 4x16 spatial tile), all 64 channels, loops t=0..31.
// Conv gates via v_wmma_f32_16x16x32_bf16 (A-fragments register-resident,
// b128 LDS reads); recurrence in f32 with v_exp/v_rcp transcendentals.
// ---------------------------------------------------------------------------

typedef __attribute__((ext_vector_type(16))) __bf16        v16bf;
typedef __attribute__((ext_vector_type(8)))  float         v8f;
typedef __attribute__((ext_vector_type(4)))  unsigned int  u32x4;

union Frag { u32x4 q[2]; v16bf v; };

__device__ __forceinline__ unsigned short f2bf(float f) {
    unsigned int u = __float_as_uint(f);
    unsigned int r = (u + 0x7FFFu + ((u >> 16) & 1u)) >> 16;   // RNE
    return (unsigned short)r;
}
__device__ __forceinline__ float sigm(float x) {
    return __builtin_amdgcn_rcpf(1.0f + __expf(-x));
}
__device__ __forceinline__ float tanh_fast(float x) {
    return 2.0f * sigm(2.0f * x) - 1.0f;
}

namespace cfg {
constexpr int B     = 4;
constexpr int CIN   = 3;
constexpr int T     = 32;
constexpr int H     = 64;
constexpr int W     = 64;
constexpr int HW    = H * W;
constexpr int COUT  = 64;
constexpr int GATES = 4 * COUT;   // 256
constexpr int K     = 54;         // CIN*2*3*3
constexpr int KP    = 64;         // padded: 2 x (16x16x32) WMMA k-steps
constexpr int TH    = 4;          // tile height
constexpr int TW    = 16;         // tile width
constexpr int NS    = TH * TW;    // 64 spatial positions / tile
constexpr int NTHR  = 256;        // 8 waves (wave32)
}

__global__ __launch_bounds__(cfg::NTHR, 1)
void convqrnn_fused(const float* __restrict__ X,
                    const float* __restrict__ Wc,
                    const float* __restrict__ bc,
                    const float* __restrict__ Wci,
                    const float* __restrict__ Wcf,
                    const float* __restrict__ Wco,
                    float* __restrict__ out)
{
    using namespace cfg;
    extern __shared__ char smem[];
    unsigned short* lw   = (unsigned short*)smem;        // [GATES][KP] bf16 weights
    unsigned short* pim  = lw + GATES * KP;              // [NS][KP]    bf16 im2col
    float*          bias = (float*)(pim + NS * KP);      // [GATES]
    float*          eg   = bias + GATES;                 // [GATES][NS] f32 gates

    const int tid  = threadIdx.x;
    const int lane = tid & 31;            // wave32
    const int wv   = tid >> 5;            // 8 waves
    const int hi   = (lane >> 4) & 1;     // upper half-wave
    const int lr   = lane & 15;

    const int bid  = blockIdx.x;
    const int b    = bid >> 6;            // batch
    const int tile = bid & 63;            // 16 x 4 tiles
    const int y0   = (tile >> 2) * TH;
    const int x0   = (tile & 3) * TW;

    // ---- stage conv weights (bf16, K padded to 64) + bias into LDS ----
    for (int idx = tid; idx < GATES * KP; idx += NTHR) {
        const int m = idx >> 6, k = idx & 63;
        const float v = (k < K) ? Wc[m * K + k] : 0.0f;   // Wconv flat: m*54 + k
        lw[idx] = f2bf(v);
    }
    for (int m = tid; m < GATES; m += NTHR) bias[m] = bc[m];

    // ---- per-thread recurrence slots: 1 pixel x 16 channels ----
    const int s   = tid & 63;             // spatial slot, s = ty*16+tx
    const int cb  = (tid >> 6) << 4;      // channel base (16 channels)
    const int sp  = (y0 + (s >> 4)) * W + (x0 + (s & 15));

    float wci[16], wcf[16], wco[16], Cst[16];
    #pragma unroll
    for (int j = 0; j < 16; ++j) {
        const int c = cb + j;
        wci[j] = Wci[c * HW + sp];
        wcf[j] = Wcf[c * HW + sp];
        wco[j] = Wco[c * HW + sp];
        Cst[j] = 0.0f;
    }

    // ---- precompute t-invariant im2col gather offsets ----
    // thread fills pim[s][k] for k = k0 + 4*i, i = 0..15
    const int    k0 = tid >> 6;                     // 0..3
    const float* Xb = X + (size_t)b * CIN * T * HW;
    int offs[16], tmin[16];
    #pragma unroll
    for (int i = 0; i < 16; ++i) {
        const int k = k0 + i * 4;
        int off = 0, tm = 1000;                     // 1000 => never valid
        if (k < K) {
            const int ci = k / 18;  int r = k - ci * 18;
            const int dt = r / 9;   r -= dt * 9;
            const int dy = r / 3;   const int dx = r - dy * 3;
            const int yy = y0 + (s >> 4) + dy - 1;  // H pad +-1
            const int xx = x0 + (s & 15) + dx - 1;  // W pad +-1
            if (yy >= 0 && yy < H && xx >= 0 && xx < W) {
                off = (ci * T + dt - 1) * HW + yy * W + xx;
                tm  = 1 - dt;                       // causal front pad: need t+dt-1 >= 0
            }
        }
        offs[i] = off;  tmin[i] = tm;
    }
    __syncthreads();

    // ---- hoist A fragments (weights) + bias into registers (t-invariant) ----
    Frag  aF[2][2];                                 // [mi][kstep]
    float biasR[2][8];
    #pragma unroll
    for (int mi = 0; mi < 2; ++mi) {
        const int m0 = (wv * 2 + mi) * 16;
        const u32x4* arow4 = (const u32x4*)(lw + (m0 + lr) * KP);  // 8 x u32x4 / row
        #pragma unroll
        for (int ks = 0; ks < 2; ++ks) {
            aF[mi][ks].q[0] = arow4[ks * 4 + hi];        // dwords ks*16 + hi*4 + 0..3
            aF[mi][ks].q[1] = arow4[ks * 4 + 2 + hi];    // dwords ks*16 + 8 + hi*4 + 0..3
        }
        #pragma unroll
        for (int r = 0; r < 8; ++r) biasR[mi][r] = bias[m0 + r + hi * 8];
    }

    for (int t = 0; t < T; ++t) {
        // ---- im2col fill: 16 predicated loads + bf16 convert per thread ----
        unsigned short* prow = pim + s * KP + k0;
        #pragma unroll
        for (int i = 0; i < 16; ++i) {
            float v = 0.0f;
            if (t >= tmin[i]) {
                v = Xb[offs[i] + t * HW];
                if (t + 1 < T) __builtin_prefetch(&Xb[offs[i] + (t + 1) * HW], 0, 3);
            }
            prow[i * 4] = f2bf(v);
        }
        __syncthreads();

        // ---- gates = W(256xK) * patchT(Kx64) + bias via WMMA bf16 ----
        #pragma unroll
        for (int n = 0; n < 4; ++n) {
            Frag b0, b1;
            const u32x4* brow4 = (const u32x4*)(pim + (n * 16 + lr) * KP);
            b0.q[0] = brow4[hi];          b0.q[1] = brow4[2 + hi];
            b1.q[0] = brow4[4 + hi];      b1.q[1] = brow4[6 + hi];
            #pragma unroll
            for (int mi = 0; mi < 2; ++mi) {
                const int m0 = (wv * 2 + mi) * 16;
                v8f c;
                #pragma unroll
                for (int r = 0; r < 8; ++r) c[r] = biasR[mi][r];
                c = __builtin_amdgcn_wmma_f32_16x16x32_bf16(
                        false, aF[mi][0].v, false, b0.v, (short)0, c, false, false);
                c = __builtin_amdgcn_wmma_f32_16x16x32_bf16(
                        false, aF[mi][1].v, false, b1.v, (short)0, c, false, false);
                #pragma unroll
                for (int r = 0; r < 8; ++r)
                    eg[(m0 + r + hi * 8) * NS + n * 16 + lr] = c[r];
            }
        }
        __syncthreads();

        // ---- peephole LSTM cell update (f32, trans ops) ----
        #pragma unroll
        for (int j = 0; j < 16; ++j) {
            const int c = cb + j;
            const float i_ = eg[(c           ) * NS + s];
            const float f_ = eg[(COUT     + c) * NS + s];
            const float g_ = eg[(2 * COUT + c) * NS + s];
            const float o_ = eg[(3 * COUT + c) * NS + s];
            const float Cp = Cst[j];
            const float iG = sigm(i_ + wci[j] * Cp);
            const float fG = sigm(f_ + wcf[j] * Cp);
            const float Cn = fG * Cp + iG * tanh_fast(g_);
            const float oG = sigm(o_ + wco[j] * Cn);
            const float Hn = oG * tanh_fast(Cn);
            Cst[j] = Cn;
            out[((b * COUT + c) * T + t) * HW + sp] = Hn;
        }
        __syncthreads();   // protect eg before next step's WMMA overwrite
    }
}

extern "C" void kernel_launch(void* const* d_in, const int* in_sizes, int n_in,
                              void* d_out, int out_size, void* d_ws, size_t ws_size,
                              hipStream_t stream) {
    using namespace cfg;
    const float* X   = (const float*)d_in[0];
    const float* Wc  = (const float*)d_in[1];
    const float* bc  = (const float*)d_in[2];
    const float* Wci = (const float*)d_in[3];
    const float* Wcf = (const float*)d_in[4];
    const float* Wco = (const float*)d_in[5];
    float* out = (float*)d_out;

    const size_t shmem = (size_t)(GATES * KP + NS * KP) * sizeof(unsigned short)
                       + (size_t)(GATES + GATES * NS) * sizeof(float);   // ~105 KB
    static bool attr_set = [&]() {
        hipFuncSetAttribute((const void*)convqrnn_fused,
                            hipFuncAttributeMaxDynamicSharedMemorySize, (int)shmem);
        return true;
    }();
    (void)attr_set;

    const int nblocks = B * (H / TH) * (W / TW);   // 4 * 16 * 4 = 256
    convqrnn_fused<<<dim3(nblocks), dim3(NTHR), shmem, stream>>>(
        X, Wc, bc, Wci, Wcf, Wco, out);
}
